// Attention_19550691131960
// MI455X (gfx1250) — matrix operand order repaired and checked by reading, not compiled, not run on hardware
//
#include <hip/hip_runtime.h>
#include <hip/hip_bf16.h>

// ---------------------------------------------------------------------------
// MHA block for B=4, N=2048, DIM=768, H=12, Dh=64 on gfx1250 (CDNA5).
// f16 operands + f32 WMMA accumulation (V_WMMA_F32_16X16X32_F16).
// GEMM stages the weight B-tile into LDS with the Tensor Data Mover
// (tensor_load_to_lds + s_wait_tensorcnt, double-buffered).
// ---------------------------------------------------------------------------

typedef __attribute__((ext_vector_type(16))) _Float16 v16h;
typedef __attribute__((ext_vector_type(8)))  _Float16 v8h;
typedef __attribute__((ext_vector_type(8)))  float    v8f;
typedef __attribute__((ext_vector_type(4)))  unsigned int u32x4;
typedef __attribute__((ext_vector_type(8)))  int          i32x8;
typedef __attribute__((ext_vector_type(4)))  int          i32x4;

constexpr int BATCH = 4;
constexpr int SEQ   = 2048;
constexpr int DIM   = 768;
constexpr int HEADS = 12;
constexpr int HD    = 64;
constexpr int MTOT  = BATCH * SEQ;   // 8192

static __device__ __forceinline__ v16h cat8(v8h lo, v8h hi) {
  v16h r;
#pragma unroll
  for (int i = 0; i < 8; ++i) { r[i] = lo[i]; r[8 + i] = hi[i]; }
  return r;
}

static __device__ __forceinline__ v8f wmma_f16(v16h a, v16h b, v8f c) {
  // D = A(16x32,f16) * B(32x16,f16) + C(16x16,f32)
  return __builtin_amdgcn_wmma_f32_16x16x32_f16(
      false, a, false, b, (short)0, c, false, false);
}

// Issue a TDM load of a 64-row x 32-half (64B/row) f16 tile into LDS.
// gsrc points at the tile start (row-major, row stride DIM halves).
static __device__ __forceinline__ void tdm_load_tile(const _Float16* gsrc,
                                                     unsigned lds_byte_addr) {
  const unsigned long long ga = (unsigned long long)(const void*)gsrc;
  u32x4 g0;
  g0[0] = 1u;                                   // count=1, user descriptor
  g0[1] = lds_byte_addr;                        // lds_addr
  g0[2] = (unsigned)(ga & 0xffffffffu);         // global_addr[31:0]
  g0[3] = (unsigned)((ga >> 32) & 0x1ffffffu)   // global_addr[56:32]
          | 0x80000000u;                        // type=2 ("image")
  i32x8 g1;
  g1[0] = (int)(1u << 16);                      // data_size=1 -> 2 bytes
  g1[1] = (int)(32u << 16);                     // tensor_dim0 = 32 (low16)
  g1[2] = (int)(64u << 16);                     // tensor_dim1 = 64 (low16)
  g1[3] = (int)(32u << 16);                     // tile_dim0  = 32
  g1[4] = (int)64;                              // tile_dim1  = 64
  g1[5] = (int)DIM;                             // tensor_dim0_stride = 768
  g1[6] = 0;
  g1[7] = 0;
  i32x4 gz4 = {0, 0, 0, 0};                     // D# groups 2/3 unused (2D)
  i32x8 gz8 = {0, 0, 0, 0, 0, 0, 0, 0};
  __builtin_amdgcn_tensor_load_to_lds(g0, g1, gz4, gz4, gz8, 0);
}

// ---------------------------------------------------------------------------
// Elementwise fp32 -> f16
// ---------------------------------------------------------------------------
__global__ void cvt_f32_to_f16(const float* __restrict__ in,
                               _Float16* __restrict__ out, int n) {
  int i = blockIdx.x * blockDim.x + threadIdx.x;
  if (i < n) out[i] = (_Float16)in[i];
}

// W[k][n] (768x768 fp32) -> WT[n][k] f16 (transposed so B-fragments are
// contiguous along K)
__global__ void cvt_transpose_w(const float* __restrict__ W,
                                _Float16* __restrict__ WT) {
  int o = blockIdx.x * blockDim.x + threadIdx.x;  // 768*768 threads
  int n = o / DIM;
  int k = o - n * DIM;
  WT[o] = (_Float16)W[k * DIM + n];
}

// ---------------------------------------------------------------------------
// WMMA GEMM: C[M=8192, N=768] = A[8192,768] * B[768,768], B given transposed.
// 64x64 tile per workgroup, 4 waves, each wave 2x2 WMMA tiles (32x32).
// Weight tile (64 rows x 32 halves) is staged into LDS by the TDM,
// double-buffered with a 1-deep pipeline (s_wait_tensorcnt 1).
// MODE 0: f16 out, layout [B,H,N,64]  (Q / K)
// MODE 2: f16 out, layout [B,H,64,N]  (V transposed)
// MODE 3: fp32 out + bias, layout [M, 768] (final projection)
// ---------------------------------------------------------------------------
template <int MODE>
__global__ __launch_bounds__(128) void gemm_wmma(
    const _Float16* __restrict__ A, const _Float16* __restrict__ BT,
    _Float16* __restrict__ outH, float* __restrict__ outF,
    const float* __restrict__ bias) {
  constexpr int TILES_N = DIM / 64;  // 12
  constexpr int NT = DIM / 32;       // 24 K-steps
  const int tm = blockIdx.x / TILES_N;
  const int tn = blockIdx.x - tm * TILES_N;
  const int wave = threadIdx.x >> 5;
  const int lane = threadIdx.x & 31;
  const int ln = lane & 15;
  const int kh = lane >> 4;

  const int m_base = tm * 64 + (wave >> 1) * 32;
  const int n_loc = (wave & 1) * 32;           // within the 64-wide tile
  const int n_base = tn * 64 + n_loc;          // global N base for this wave

  __shared__ __align__(64) _Float16 Bsm[2][64][32];  // 8 KB double buffer

  const _Float16* btile = BT + (size_t)(tn * 64) * DIM;  // tile row 0

  // prologue: kick off tile 0
  if (threadIdx.x == 0)
    tdm_load_tile(btile, (unsigned)(size_t)(&Bsm[0][0][0]));

  v8f c[2][2] = {};

  for (int t = 0; t < NT; ++t) {
    const int buf = t & 1;
    __syncthreads();  // previous readers of buf^1 are done; safe to refill it
    if (threadIdx.x == 0 && (t + 1) < NT)
      tdm_load_tile(btile + (t + 1) * 32,
                    (unsigned)(size_t)(&Bsm[buf ^ 1][0][0]));
    if (wave == 0) {
      if ((t + 1) < NT)
        __builtin_amdgcn_s_wait_tensorcnt(1);  // tile t landed, t+1 in flight
      else
        __builtin_amdgcn_s_wait_tensorcnt(0);
    }
    __syncthreads();  // release all waves to read buf

    const int k0 = t * 32;
    v16h a[2], b[2];
#pragma unroll
    for (int i = 0; i < 2; ++i) {
      const _Float16* ap =
          A + (size_t)(m_base + i * 16 + ln) * DIM + k0 + kh * 16;
      a[i] = cat8(*(const v8h*)ap, *(const v8h*)(ap + 8));
    }
#pragma unroll
    for (int j = 0; j < 2; ++j) {
      b[j] = *(const v16h*)(&Bsm[buf][n_loc + j * 16 + ln][kh * 16]);
    }
#pragma unroll
    for (int i = 0; i < 2; ++i)
#pragma unroll
      for (int j = 0; j < 2; ++j) c[i][j] = wmma_f16(a[i], b[j], c[i][j]);
  }

#pragma unroll
  for (int i = 0; i < 2; ++i) {
#pragma unroll
    for (int j = 0; j < 2; ++j) {
#pragma unroll
      for (int r = 0; r < 8; ++r) {
        const int row = m_base + i * 16 + kh * 8 + r;  // global M
        const int col = n_base + j * 16 + ln;          // global N
        const float v = c[i][j][r];
        if (MODE == 3) {
          outF[(size_t)row * DIM + col] = v + bias[col];
        } else {
          const int bb = row >> 11;          // row / SEQ
          const int nn = row & (SEQ - 1);    // row % SEQ
          const int hh = col >> 6;           // col / HD
          const int dd = col & (HD - 1);     // col % HD
          if (MODE == 0)
            outH[(((size_t)bb * HEADS + hh) * SEQ + nn) * HD + dd] = (_Float16)v;
          else  // MODE == 2: V transposed [B,H,HD,SEQ]
            outH[(((size_t)bb * HEADS + hh) * HD + dd) * SEQ + nn] = (_Float16)v;
        }
      }
    }
  }
}

// ---------------------------------------------------------------------------
// Flash attention: each block = one (b,h) and 64 queries; each wave owns 16
// queries and streams all 2048 keys in blocks of 32.
// Q,K: [B,H,N,64] f16. Vt: [B,H,64,N] f16. O: [B,N,768] f16.
// ---------------------------------------------------------------------------
__global__ __launch_bounds__(128) void attention_wmma(
    const _Float16* __restrict__ Q, const _Float16* __restrict__ K,
    const _Float16* __restrict__ Vt, _Float16* __restrict__ O) {
  const int bh = blockIdx.x >> 5;          // 0..47
  const int qblk = blockIdx.x & 31;        // 0..31
  const int bb = bh / HEADS;
  const int hh = bh - bb * HEADS;
  const int wave = threadIdx.x >> 5;
  const int lane = threadIdx.x & 31;
  const int ln = lane & 15;
  const int kh = lane >> 4;
  const int q0 = qblk * 64 + wave * 16;

  const _Float16* Qb = Q + (size_t)bh * SEQ * HD;
  const _Float16* Kb = K + (size_t)bh * SEQ * HD;
  const _Float16* Vb = Vt + (size_t)bh * HD * SEQ;

  __shared__ __align__(32) _Float16 Plds[4][16][32];  // per-wave P tile

  // Q A-fragments for K-dim 0..31 and 32..63 (fixed across key loop)
  const _Float16* qp = Qb + (size_t)(q0 + ln) * HD + kh * 16;
  const v16h aQ0 = cat8(*(const v8h*)qp,        *(const v8h*)(qp + 8));
  const v16h aQ1 = cat8(*(const v8h*)(qp + 32), *(const v8h*)(qp + 40));

  const float scale = 0.125f;  // 1/sqrt(64)
  float mrow[8], lrow[8];
  v8f acc[4] = {};
#pragma unroll
  for (int r = 0; r < 8; ++r) { mrow[r] = -1e30f; lrow[r] = 0.0f; }

  for (int j = 0; j < SEQ; j += 32) {
    if (j + 32 < SEQ) {  // L2 prefetch of next key/value block
      __builtin_prefetch(Kb + (size_t)(j + 32 + ln) * HD, 0, 1);
      __builtin_prefetch(Vb + (size_t)ln * SEQ + j + 32, 0, 1);
    }
    // ---- S = Q K^T for a 16x32 tile (two 16-key column tiles) ----
    const _Float16* kp0 = Kb + (size_t)(j + ln) * HD;
    const _Float16* kp1 = Kb + (size_t)(j + 16 + ln) * HD;
    v16h bK;
    bK = *(const v16h*)(kp0 + kh * 16);
    v8f s0 = wmma_f16(aQ0, bK, v8f{});
    bK = *(const v16h*)(kp0 + 32 + kh * 16);
    s0 = wmma_f16(aQ1, bK, s0);
    bK = *(const v16h*)(kp1 + kh * 16);
    v8f s1 = wmma_f16(aQ0, bK, v8f{});
    bK = *(const v16h*)(kp1 + 32 + kh * 16);
    s1 = wmma_f16(aQ1, bK, s1);

    float sc0[8], sc1[8], tmax[8];
#pragma unroll
    for (int r = 0; r < 8; ++r) {
      sc0[r] = s0[r] * scale;
      sc1[r] = s1[r] * scale;
      tmax[r] = fmaxf(sc0[r], sc1[r]);
    }
    // row max across the 16-lane group (rows live at 8*kh + r)
#pragma unroll
    for (int off = 1; off < 16; off <<= 1)
#pragma unroll
      for (int r = 0; r < 8; ++r)
        tmax[r] = fmaxf(tmax[r], __shfl_xor(tmax[r], off, 32));

    float alpha[8], p0[8], p1[8], psum[8];
#pragma unroll
    for (int r = 0; r < 8; ++r) {
      const float mn = fmaxf(mrow[r], tmax[r]);
      alpha[r] = __expf(mrow[r] - mn);
      mrow[r] = mn;
      p0[r] = __expf(sc0[r] - mn);
      p1[r] = __expf(sc1[r] - mn);
      psum[r] = p0[r] + p1[r];
    }
#pragma unroll
    for (int off = 1; off < 16; off <<= 1)
#pragma unroll
      for (int r = 0; r < 8; ++r) psum[r] += __shfl_xor(psum[r], off, 32);
#pragma unroll
    for (int r = 0; r < 8; ++r) lrow[r] = lrow[r] * alpha[r] + psum[r];
#pragma unroll
    for (int dt = 0; dt < 4; ++dt)
#pragma unroll
      for (int r = 0; r < 8; ++r) acc[dt][r] *= alpha[r];

    // ---- transpose P through LDS into A-fragment layout ----
#pragma unroll
    for (int r = 0; r < 8; ++r) {
      Plds[wave][kh * 8 + r][ln] = (_Float16)p0[r];
      Plds[wave][kh * 8 + r][16 + ln] = (_Float16)p1[r];
    }
    // same-wave DS ops are in-order; compiler inserts s_wait_dscnt
    const v16h aP = cat8(*(const v8h*)(&Plds[wave][ln][kh * 16]),
                         *(const v8h*)(&Plds[wave][ln][kh * 16 + 8]));

    // ---- O += P * V (4 d-tiles of 16) ----
#pragma unroll
    for (int dt = 0; dt < 4; ++dt) {
      const v16h bV =
          *(const v16h*)(Vb + (size_t)(dt * 16 + ln) * SEQ + j + kh * 16);
      acc[dt] = wmma_f16(aP, bV, acc[dt]);
    }
  }

  // ---- finalize and store O[b, n, h*64 + d] ----
#pragma unroll
  for (int dt = 0; dt < 4; ++dt) {
#pragma unroll
    for (int r = 0; r < 8; ++r) {
      const float o = acc[dt][r] / lrow[r];
      const int row = q0 + kh * 8 + r;
      const int col = hh * HD + dt * 16 + ln;
      O[((size_t)bb * SEQ + row) * DIM + col] = (_Float16)o;
    }
  }
}

// ---------------------------------------------------------------------------
// Host launcher
// ---------------------------------------------------------------------------
extern "C" void kernel_launch(void* const* d_in, const int* in_sizes, int n_in,
                              void* d_out, int out_size, void* d_ws,
                              size_t ws_size, hipStream_t stream) {
  const float* x  = (const float*)d_in[0];
  const float* Wq = (const float*)d_in[1];
  const float* Wk = (const float*)d_in[2];
  const float* Wv = (const float*)d_in[3];
  const float* Wp = (const float*)d_in[4];
  const float* bp = (const float*)d_in[5];

  char* ws = (char*)d_ws;
  size_t off = 0;
  auto carve = [&](size_t bytes) -> char* {
    char* p = ws + off;
    off = (off + bytes + 255) & ~(size_t)255;
    return p;
  };

  const size_t nX = (size_t)MTOT * DIM;       // 6,291,456
  const size_t nW = (size_t)DIM * DIM;        //   589,824

  _Float16* xh  = (_Float16*)carve(nX * 2);
  _Float16* wqT = (_Float16*)carve(nW * 2);
  _Float16* wkT = (_Float16*)carve(nW * 2);
  _Float16* wvT = (_Float16*)carve(nW * 2);
  _Float16* wpT = (_Float16*)carve(nW * 2);
  _Float16* Qh  = (_Float16*)carve(nX * 2);
  _Float16* Kh  = (_Float16*)carve(nX * 2);
  _Float16* Vth = (_Float16*)carve(nX * 2);
  _Float16* Oh  = (_Float16*)carve(nX * 2);
  (void)ws_size;

  // 1. precision conversion + weight transpose
  cvt_f32_to_f16<<<(int)(nX / 256), 256, 0, stream>>>(x, xh, (int)nX);
  cvt_transpose_w<<<(int)(nW / 256), 256, 0, stream>>>(Wq, wqT);
  cvt_transpose_w<<<(int)(nW / 256), 256, 0, stream>>>(Wk, wkT);
  cvt_transpose_w<<<(int)(nW / 256), 256, 0, stream>>>(Wv, wvT);
  cvt_transpose_w<<<(int)(nW / 256), 256, 0, stream>>>(Wp, wpT);

  // 2. QKV projections (WMMA + TDM-staged weight tiles)
  const int gemmGrid = (MTOT / 64) * (DIM / 64);  // 1536
  gemm_wmma<0><<<gemmGrid, 128, 0, stream>>>(xh, wqT, Qh, nullptr, nullptr);
  gemm_wmma<0><<<gemmGrid, 128, 0, stream>>>(xh, wkT, Kh, nullptr, nullptr);
  gemm_wmma<2><<<gemmGrid, 128, 0, stream>>>(xh, wvT, Vth, nullptr, nullptr);

  // 3. flash attention (WMMA)
  attention_wmma<<<BATCH * HEADS * (SEQ / 64), 128, 0, stream>>>(Qh, Kh, Vth, Oh);

  // 4. output projection + bias (WMMA, fp32 out)
  gemm_wmma<3><<<gemmGrid, 128, 0, stream>>>(Oh, wpT, nullptr, (float*)d_out, bp);
}
